// HierarchicalMamba_7370163880429
// MI455X (gfx1250) — compile-verified, matrix operand order checked by reading, not run on hardware
//
#include <hip/hip_runtime.h>
#include <hip/hip_bf16.h>
#include <math.h>

// Problem dims (match reference)
#define B_  2
#define L_  1024
#define DM_ 768
#define DI_ 1536
#define DS_ 16
#define DC_ 4
#define DR_ 48
#define M_  (B_ * L_)          // 2048 rows of activations
#define DBL_ (DR_ + 2 * DS_)   // 80 (logical)
#define DBLP_ 128              // padded N / row stride for the x_proj output

typedef __attribute__((ext_vector_type(8)))  __bf16 v8bf;
typedef __attribute__((ext_vector_type(16))) __bf16 v16bf;
typedef __attribute__((ext_vector_type(8)))  float  v8f;

// ---------------------------------------------------------------------------
// f32 -> bf16 convert / zero fill
// ---------------------------------------------------------------------------
__global__ __launch_bounds__(256) void cvt_f32_bf16(const float* __restrict__ src,
                                                    __bf16* __restrict__ dst, int n) {
  int i = blockIdx.x * 256 + threadIdx.x;
  if (i < n) dst[i] = (__bf16)src[i];
}

__global__ __launch_bounds__(256) void fill_zero_bf16(__bf16* __restrict__ dst, int n) {
  int i = blockIdx.x * 256 + threadIdx.x;
  if (i < n) dst[i] = (__bf16)0.0f;
}

// f32 (rows x cols, row stride `stride`) -> bf16 (rows x dcols) zero-padded cols
__global__ __launch_bounds__(256) void pad_cvt_f32_bf16(const float* __restrict__ src,
                                                        __bf16* __restrict__ dst,
                                                        int rows, int cols, int stride, int dcols) {
  int i = blockIdx.x * 256 + threadIdx.x;
  if (i >= rows * dcols) return;
  int r = i / dcols, c = i - r * dcols;
  dst[i] = (c < cols) ? (__bf16)src[(size_t)r * stride + c] : (__bf16)0.0f;
}

// ---------------------------------------------------------------------------
// Register-blocked WMMA GEMM: C[M,N] = alpha * A[M,K] @ W[N,K]^T + beta * C
// A, W bf16 row-major; C f32 row-major (ldc = N).
// One wave computes a 32x64 macro-tile = 2(M) x 4(N) tiles of 16x16 via
// v_wmma_f32_16x16x32_bf16. Per K-step of 32: 2 A-frags + 4 B-frags (12 b128
// loads) feed 8 WMMAs.
// CDNA5 16-bit A layout: lane = row (lane&15); lanes 0-15 hold K=[0..8)+[16..24),
// lanes 16-31 hold K=[8..16)+[24..32) per K-step. W is [N,K] row-major
// (column-major B) -> identical contiguous-row fragment loads.
// Requires: M % 256 == 0 (8 waves of 32 rows per block.y), N % 64 == 0, K % 32 == 0.
// ---------------------------------------------------------------------------
__global__ __launch_bounds__(256) void wmma_gemm_bf16(const __bf16* __restrict__ A,
                                                      const __bf16* __restrict__ W,
                                                      float* __restrict__ C,
                                                      int M, int N, int K,
                                                      float alpha, float beta) {
  const int lane = threadIdx.x & 31;
  const int wave = threadIdx.x >> 5;
  const int m0 = (blockIdx.y * 8 + wave) * 32;   // macro-tile M origin
  const int n0 = blockIdx.x * 64;                // macro-tile N origin

  const int row = lane & 15;
  const int klo = (lane & 16) ? 8 : 0;

  const __bf16* __restrict__ arow[2];
  const __bf16* __restrict__ wrow[4];
#pragma unroll
  for (int mi = 0; mi < 2; ++mi)
    arow[mi] = A + (size_t)(m0 + mi * 16 + row) * K + klo;
#pragma unroll
  for (int ni = 0; ni < 4; ++ni)
    wrow[ni] = W + (size_t)(n0 + ni * 16 + row) * K + klo;

  v8f acc[2][4];
#pragma unroll
  for (int mi = 0; mi < 2; ++mi)
#pragma unroll
    for (int ni = 0; ni < 4; ++ni) acc[mi][ni] = (v8f){};

  for (int k0 = 0; k0 < K; k0 += 32) {
    if (k0 + 32 < K) {
#pragma unroll
      for (int mi = 0; mi < 2; ++mi) __builtin_prefetch(arow[mi] + k0 + 32, 0, 1);
#pragma unroll
      for (int ni = 0; ni < 4; ++ni) __builtin_prefetch(wrow[ni] + k0 + 32, 0, 1);
    }
    v16bf af[2], bf[4];
#pragma unroll
    for (int mi = 0; mi < 2; ++mi) {
      v8bf a0 = *(const v8bf*)(arow[mi] + k0);
      v8bf a1 = *(const v8bf*)(arow[mi] + k0 + 16);
      af[mi] = __builtin_shufflevector(a0, a1, 0, 1, 2, 3, 4, 5, 6, 7,
                                               8, 9, 10, 11, 12, 13, 14, 15);
    }
#pragma unroll
    for (int ni = 0; ni < 4; ++ni) {
      v8bf b0 = *(const v8bf*)(wrow[ni] + k0);
      v8bf b1 = *(const v8bf*)(wrow[ni] + k0 + 16);
      bf[ni] = __builtin_shufflevector(b0, b1, 0, 1, 2, 3, 4, 5, 6, 7,
                                               8, 9, 10, 11, 12, 13, 14, 15);
    }
#pragma unroll
    for (int mi = 0; mi < 2; ++mi)
#pragma unroll
      for (int ni = 0; ni < 4; ++ni)
        acc[mi][ni] = __builtin_amdgcn_wmma_f32_16x16x32_bf16(
            false, af[mi], false, bf[ni], (short)0, acc[mi][ni], false, false);
  }

  // D layout: lane -> col = lane&15; VGPR v -> row = v + (lane<16 ? 0 : 8)
  const int col = lane & 15;
  const int mb  = (lane & 16) ? 8 : 0;
#pragma unroll
  for (int mi = 0; mi < 2; ++mi) {
#pragma unroll
    for (int ni = 0; ni < 4; ++ni) {
      float* __restrict__ cbase =
          C + (size_t)(m0 + mi * 16 + mb) * N + n0 + ni * 16 + col;
#pragma unroll
      for (int v = 0; v < 8; ++v) {
        float val = alpha * acc[mi][ni][v];
        if (beta != 0.0f) val += beta * cbase[(size_t)v * N];
        cbase[(size_t)v * N] = val;
      }
    }
  }
}

// ---------------------------------------------------------------------------
// depthwise causal conv (K=4) + bias + SiLU on the x half of xz
// xz: [B*L, 2*DI] f32 ; outputs xconv f32 and bf16
// ---------------------------------------------------------------------------
__global__ __launch_bounds__(256) void conv_silu(const float* __restrict__ xz,
                                                 const float* __restrict__ cw,
                                                 const float* __restrict__ cb,
                                                 float* __restrict__ xf,
                                                 __bf16* __restrict__ xh) {
  int idx = blockIdx.x * 256 + threadIdx.x;      // over B*L*DI
  if (idx >= M_ * DI_) return;
  int d  = idx % DI_;
  int bl = idx / DI_;
  int l  = bl % L_;
  int b  = bl / L_;
  float acc = cb[d];
#pragma unroll
  for (int j = 0; j < DC_; ++j) {
    int ls = l - (DC_ - 1) + j;
    if (ls >= 0)
      acc += xz[((size_t)(b * L_ + ls)) * (2 * DI_) + d] * cw[d * DC_ + j];
  }
  float s = acc / (1.0f + expf(-acc));           // SiLU
  xf[idx] = s;
  xh[idx] = (__bf16)s;
}

// ---------------------------------------------------------------------------
// Selective scan. One thread per (b,d) channel: 16-state recurrence in regs,
// sequential over L. B/C vectors (16 each) broadcast via LDS per timestep.
// Fused: delta = softplus(dt_raw*scale + dtb*scale), y = (h.C) + x*D,
// y *= SiLU(z), store bf16 for the out_proj GEMM.
// ---------------------------------------------------------------------------
__global__ __launch_bounds__(256) void selective_scan(const float* __restrict__ dbl,
                                                      const float* __restrict__ dt_raw,
                                                      const float* __restrict__ dtb,
                                                      const float* __restrict__ A_log,
                                                      const float* __restrict__ Dv,
                                                      const float* __restrict__ xz,
                                                      const float* __restrict__ xconv,
                                                      __bf16* __restrict__ ybf,
                                                      float scale) {
  const int chunks = DI_ / 256;                  // 6
  const int b = blockIdx.x / chunks;
  const int d = (blockIdx.x % chunks) * 256 + threadIdx.x;

  __shared__ float sB[DS_];
  __shared__ float sC[DS_];

  float Av[DS_];
#pragma unroll
  for (int n = 0; n < DS_; ++n) Av[n] = -expf(A_log[d * DS_ + n]);
  const float Dp   = Dv[d];
  const float bias = dtb[d] * scale;

  float h[DS_];
#pragma unroll
  for (int n = 0; n < DS_; ++n) h[n] = 0.0f;

  for (int l = 0; l < L_; ++l) {
    const size_t m = (size_t)b * L_ + l;
    if (threadIdx.x < 2 * DS_) {
      int t = threadIdx.x;
      float v = dbl[m * DBLP_ + DR_ + t];
      if (t < DS_) sB[t] = v; else sC[t - DS_] = v;
    }
    __syncthreads();

    float dr    = dt_raw[m * DI_ + d] * scale + bias;
    float delta = (dr > 20.0f) ? dr : log1pf(expf(dr));
    float x     = xconv[m * DI_ + d];
    float dx    = delta * x;
    float y     = 0.0f;
#pragma unroll
    for (int n = 0; n < DS_; ++n) {
      h[n] = expf(delta * Av[n]) * h[n] + dx * sB[n];
      y += h[n] * sC[n];
    }
    y += x * Dp;
    float z = xz[m * (2 * DI_) + DI_ + d];
    y *= z / (1.0f + expf(-z));
    ybf[m * DI_ + d] = (__bf16)y;
    __syncthreads();                             // before next timestep's B/C load
  }
}

// ---------------------------------------------------------------------------
// Host: 4 branches, sequential, reusing workspace; dt_raw persists from br 0.
// ---------------------------------------------------------------------------
extern "C" void kernel_launch(void* const* d_in, const int* in_sizes, int n_in,
                              void* d_out, int out_size, void* d_ws, size_t ws_size,
                              hipStream_t stream) {
  const float* h    = (const float*)d_in[0];   // (B,L,DM)
  const float* iw   = (const float*)d_in[1];   // (4, 2*DI, DM)
  const float* cw   = (const float*)d_in[2];   // (4, DI, DC)
  const float* cb   = (const float*)d_in[3];   // (4, DI)
  const float* xw   = (const float*)d_in[4];   // (4, DR+2*DS, DI)
  const float* dtw  = (const float*)d_in[5];   // (4, DI, DR)
  const float* dtb  = (const float*)d_in[6];   // (4, DI)
  const float* alog = (const float*)d_in[7];   // (4, DI, DS)
  const float* Dv   = (const float*)d_in[8];   // (4, DI)
  const float* ow   = (const float*)d_in[9];   // (4, DM, DI)
  float* out = (float*)d_out;                  // (B,L,DM)

  char* p = (char*)d_ws;
  auto take = [&](size_t bytes) -> void* {
    void* r = (void*)p;
    p += (bytes + 255) & ~(size_t)255;
    return r;
  };
  __bf16* h_bf   = (__bf16*)take((size_t)M_ * DM_ * 2);
  __bf16* w_in   = (__bf16*)take((size_t)2 * DI_ * DM_ * 2);
  __bf16* w_x    = (__bf16*)take((size_t)DBLP_ * DI_ * 2);   // rows 80..127 zero
  __bf16* w_dt   = (__bf16*)take((size_t)DI_ * 64 * 2);      // K padded 48->64
  __bf16* w_out  = (__bf16*)take((size_t)DM_ * DI_ * 2);
  float*  xz     = (float*) take((size_t)M_ * 2 * DI_ * 4);
  float*  xcf    = (float*) take((size_t)M_ * DI_ * 4);
  __bf16* xch    = (__bf16*)take((size_t)M_ * DI_ * 2);
  float*  dbl    = (float*) take((size_t)M_ * DBLP_ * 4);    // row stride 128
  __bf16* dtr    = (__bf16*)take((size_t)M_ * 64 * 2);       // K padded 48->64
  float*  dt_raw = (float*) take((size_t)M_ * DI_ * 4);
  __bf16* ybf    = (__bf16*)take((size_t)M_ * DI_ * 2);

  auto cvt = [&](const float* s, __bf16* d, int n) {
    cvt_f32_bf16<<<(n + 255) / 256, 256, 0, stream>>>(s, d, n);
  };
  // grid for a register-blocked GEMM: x = N/64 macro-cols, y = M/256 (8 waves * 32 rows)
  auto gemm_grid = [&](int N) { return dim3((unsigned)(N / 64), (unsigned)(M_ / 256)); };

  // hidden states once
  cvt(h, h_bf, M_ * DM_);
  // dt weight (branch 0 only is used), zero-padded K: 48 -> 64
  {
    int n = DI_ * 64;
    pad_cvt_f32_bf16<<<(n + 255) / 256, 256, 0, stream>>>(dtw, w_dt, DI_, DR_, DR_, 64);
  }

  const float scales[4] = {1.0f, 2.0f, 4.0f, 8.0f};

  for (int br = 0; br < 4; ++br) {
    // per-branch weight conversions
    cvt(iw + (size_t)br * 2 * DI_ * DM_, w_in, 2 * DI_ * DM_);
    cvt(ow + (size_t)br * DM_ * DI_,     w_out, DM_ * DI_);
    // x_proj weight: convert 80 rows, zero rows 80..127 so padded N cols = 0
    cvt(xw + (size_t)br * DBL_ * DI_, w_x, DBL_ * DI_);
    {
      int n = (DBLP_ - DBL_) * DI_;
      fill_zero_bf16<<<(n + 255) / 256, 256, 0, stream>>>(w_x + (size_t)DBL_ * DI_, n);
    }

    // xz = h @ in_w^T : M=2048, N=3072, K=768
    wmma_gemm_bf16<<<gemm_grid(2 * DI_), 256, 0, stream>>>(
        h_bf, w_in, xz, M_, 2 * DI_, DM_, 1.0f, 0.0f);

    // causal conv + SiLU on x half
    {
      int n = M_ * DI_;
      conv_silu<<<(n + 255) / 256, 256, 0, stream>>>(
          xz, cw + (size_t)br * DI_ * DC_, cb + (size_t)br * DI_, xcf, xch);
    }

    // dbl = xconv @ x_w^T : M=2048, N=128 (padded from 80), K=1536
    wmma_gemm_bf16<<<gemm_grid(DBLP_), 256, 0, stream>>>(
        xch, w_x, dbl, M_, DBLP_, DI_, 1.0f, 0.0f);

    if (br == 0) {
      // dt_raw = dbl[:, :48] @ dt_w^T : pad K 48->64, M=2048, N=1536, K=64
      int n = M_ * 64;
      pad_cvt_f32_bf16<<<(n + 255) / 256, 256, 0, stream>>>(dbl, dtr, M_, DR_, DBLP_, 64);
      wmma_gemm_bf16<<<gemm_grid(DI_), 256, 0, stream>>>(
          dtr, w_dt, dt_raw, M_, DI_, 64, 1.0f, 0.0f);
    }

    // selective scan (+ gate + bf16 store)
    selective_scan<<<dim3(B_ * (DI_ / 256)), 256, 0, stream>>>(
        dbl, dt_raw, dtb /* branch-0 bias, scaled */,
        alog + (size_t)br * DI_ * DS_, Dv + (size_t)br * DI_,
        xz, xcf, ybf, scales[br]);

    // out += 0.25 * (y @ out_w^T) : M=2048, N=768, K=1536
    wmma_gemm_bf16<<<gemm_grid(DM_), 256, 0, stream>>>(
        ybf, w_out, out, M_, DM_, DI_, 0.25f, br == 0 ? 0.0f : 1.0f);
  }
}
